// ChiENN_19421842112971
// MI455X (gfx1250) — matrix operand order repaired and checked by reading, not compiled, not run on hardware
//
#include <hip/hip_runtime.h>

// ============================================================================
// ChiENN fused attention for MI455X (gfx1250, wave32, WMMA bf16 16x16x32).
//
// Roofline: restructured algorithm (Q/V projections folded through the
// attention) does ~28 GFLOP on ~550 MB of traffic -> memory bound (~24us
// floor @ 23.3 TB/s). ALL matrix work incl. the attention context reduction
// runs on v_wmma bf16/f32-accum; weights converted to bf16 once (768KB ws,
// L2-resident); every input byte read exactly once. f32->bf16 uses the
// native cvt path so staging hides under the HBM stream.
// ============================================================================

typedef unsigned short u16;
typedef __attribute__((ext_vector_type(16))) __bf16 v16bf;
typedef __attribute__((ext_vector_type(2)))  __bf16 v2bf;
typedef __attribute__((ext_vector_type(8)))  float  v8f;
typedef __attribute__((ext_vector_type(8)))  u16    ush8;

#define L_DIM 8
#define KSLOT 18          // 2L+2 message slots
#define NT 16             // nodes per workgroup
#define MS 264            // padded row stride (bf16 elems): 528B -> conflict-free A frags
#define NODE_STRIDE (KSLOT*MS)   // 4752
#define CTX_STRIDE  (8*MS)       // 2112
#define PB_HS 48                 // p-buffer per-head stride (96B, 16B aligned)
#define PB_NS (8*PB_HS)          // 384

// ---- static LDS carve-up (316 KB of the WGP's 320 KB) ----------------------
#define MSGB_OFF 0               // u16 [16][18][264]      152064 B
#define ZB_OFF   152064          // u16 [16][256][8]        65536 B
#define CTXB_OFF 217600          // u16 [16][8][264]        67584 B
#define SCP_OFF  285184          // f32 [16][18][8]          9216 B (raw scores)
#define KKB_OFF  294400          // u16 [16][264]            8448 B (kvec bf16)
#define XA_OFF   302848          // u16 [16][264]            8448 B (x rows / outv)
#define PB_OFF   311296          // u16 [16][8][48]         12288 B (p bf16, A-layout)
#define MB_OFF   323584          // u32 [16] mask bits         64 B
#define SMEM_TOTAL 323648

// Native f32->bf16 (RNE); clang lowers to the HW cvt op on gfx1250.
__device__ inline u16 f2bf(float f) {
  union { __bf16 b; u16 u; } x;
  x.b = (__bf16)f;
  return x.u;
}
// Packed pair -> one dword (candidate for v_cvt_pk_bf16_f32).
__device__ inline unsigned int pk2bf(float x, float y) {
  union { v2bf v; unsigned int u; } r;
  r.v[0] = (__bf16)x;
  r.v[1] = (__bf16)y;
  return r.u;
}
__device__ inline v8f zero8() { v8f z = {0.f,0.f,0.f,0.f,0.f,0.f,0.f,0.f}; return z; }

// A fragment (16xK=32 bf16). ISA layout: lane m=l&15, half=l>>4;
// packed K = {half*8 .. half*8+7, 16+half*8 .. 16+half*8+7}.
__device__ inline v16bf ldA(const u16* base, int rs, int lane) {
  int m = lane & 15, h = lane >> 4;
  const u16* p = base + m * rs + h * 8;
  union { v16bf v; ush8 u[2]; } x;
  x.u[0] = *(const ush8*)(p);
  x.u[1] = *(const ush8*)(p + 16);
  return x.v;
}
// A fragment with only nvalid rows present (rest zero).
__device__ inline v16bf ldA_g(const u16* base, int rs, int lane, int nvalid) {
  int m = lane & 15, h = lane >> 4;
  union { v16bf v; ush8 u[2]; } x;
  ush8 z = {0,0,0,0,0,0,0,0};
  x.u[0] = z; x.u[1] = z;
  if (m < nvalid) {
    const u16* p = base + m * rs + h * 8;
    x.u[0] = *(const ush8*)(p);
    x.u[1] = *(const ush8*)(p + 16);
  }
  return x.v;
}
// B fragment (K=32 x N=16 bf16): lane col n=l&15, K in A's packed order.
__device__ inline v16bf ldBg(const u16* p, int rs, int lane) {
  int n = lane & 15, h = lane >> 4;
  union { v16bf v; u16 u[16]; } x;
#pragma unroll
  for (int s = 0; s < 16; ++s) {
    int k = s < 8 ? h * 8 + s : 8 + h * 8 + s;
    x.u[s] = p[k * rs + n];
  }
  return x.v;
}
// B fragment from strided rows with only nvalid K rows (rest zero).
__device__ inline v16bf ldB_rows(const u16* base, int rs, int lane, int nvalid) {
  int n = lane & 15, h = lane >> 4;
  union { v16bf v; u16 u[16]; } x;
#pragma unroll
  for (int s = 0; s < 16; ++s) {
    int k = s < 8 ? h * 8 + s : 8 + h * 8 + s;
    x.u[s] = k < nvalid ? base[k * rs + n] : (u16)0;
  }
  return x.v;
}
// B fragment with only 8 valid N columns (heads), upper 8 zero.
__device__ inline v16bf ldB_half(const u16* p, int lane) {
  int n = lane & 15, h = lane >> 4;
  union { v16bf v; u16 u[16]; } x;
#pragma unroll
  for (int s = 0; s < 16; ++s) {
    int k = s < 8 ? h * 8 + s : 8 + h * 8 + s;
    x.u[s] = n < 8 ? p[k * 8 + n] : (u16)0;
  }
  return x.v;
}

__device__ inline v8f wmma_bf16(v16bf a, v16bf b, v8f c) {
  return __builtin_amdgcn_wmma_f32_16x16x32_bf16(false, a, false, b, (short)0, c,
                                                 false, false);
}

// 16 rows x 32 cols of C = A(16x256) @ B(256xNcols), two 16-wide col tiles.
__device__ inline void gemm_16x32(const u16* A, int Ars, const u16* Bg, int col0,
                                  int lane, v8f& a0, v8f& a1) {
#pragma unroll
  for (int kk = 0; kk < 256; kk += 32) {
    v16bf af = ldA(A + kk, Ars, lane);
    v16bf b0 = ldBg(Bg + (size_t)kk * 256 + col0, 256, lane);
    v16bf b1 = ldBg(Bg + (size_t)kk * 256 + col0 + 16, 256, lane);
    a0 = wmma_bf16(af, b0, a0);
    a1 = wmma_bf16(af, b1, a1);
  }
}

// C/D layout: elem r of v8f -> row = r + (lane>>4)*8, col = lane&15.
__device__ inline void storeD_bf16(u16* base, int rowStride, int col0, int lane,
                                   const v8f& a0, const v8f& a1) {
  int half = lane >> 4, cl = lane & 15;
#pragma unroll
  for (int r = 0; r < 8; ++r) {
    int row = r + half * 8;
    base[row * rowStride + col0 + cl]      = f2bf(a0[r]);
    base[row * rowStride + col0 + 16 + cl] = f2bf(a1[r]);
  }
}

// ============================================================================
// Kernel 0: convert the six 256x256 f32 weights to bf16 in ws.
// Slabs (65536 u16 each): 0 WselfT 1 WparT 2 WkT 3 Wq(plain) 4 WvT 5 WfinalT
// ============================================================================
__global__ void wconv(const float* __restrict__ wself, const float* __restrict__ wpar,
                      const float* __restrict__ wk,   const float* __restrict__ wq,
                      const float* __restrict__ wv,   const float* __restrict__ wfin,
                      u16* __restrict__ dst) {
  int m = blockIdx.y, i = blockIdx.x, j = threadIdx.x;
  const float* src; bool tr = true;
  switch (m) {
    case 0: src = wself; break;
    case 1: src = wpar;  break;
    case 2: src = wk;    break;
    case 3: src = wq; tr = false; break;   // Z-GEMM B uses Wq rows directly
    case 4: src = wv;    break;
    default: src = wfin; break;
  }
  float v = src[i * 256 + j];
  u16* d = dst + (size_t)m * 65536;
  if (tr) d[j * 256 + i] = f2bf(v);
  else    d[i * 256 + j] = f2bf(v);
}

// ============================================================================
// Kernel 1: fully fused ChiENN aggregate, one 16-node tile per workgroup.
// ============================================================================
__global__ __launch_bounds__(256, 1)
void chienn_fused(const float* __restrict__ bx, const int* __restrict__ pidx,
                  const float* __restrict__ ccw, const unsigned char* __restrict__ ccwm,
                  const float* __restrict__ cw,  const unsigned char* __restrict__ cwm,
                  const u16* __restrict__ wbf, float* __restrict__ outp, int Ntot) {
  __shared__ __align__(16) unsigned char smem[SMEM_TOTAL];
  u16*   msgB  = (u16*)(smem + MSGB_OFF);
  u16*   Zb    = (u16*)(smem + ZB_OFF);
  u16*   ctxb  = (u16*)(smem + CTXB_OFF);
  float* scp   = (float*)(smem + SCP_OFF);
  u16*   kkb   = (u16*)(smem + KKB_OFF);
  u16*   xa    = (u16*)(smem + XA_OFF);
  u16*   pb    = (u16*)(smem + PB_OFF);
  unsigned int* maskb = (unsigned int*)(smem + MB_OFF);

  const u16* wSelfT = wbf + 0 * 65536;
  const u16* wParT  = wbf + 1 * 65536;
  const u16* wKT    = wbf + 2 * 65536;
  const u16* wQ     = wbf + 3 * 65536;
  const u16* wVT    = wbf + 4 * 65536;
  const u16* wFinT  = wbf + 5 * 65536;

  const int tid = threadIdx.x;
  const int w = tid >> 5, lane = tid & 31;
  const int n0 = blockIdx.x * NT;

  // ---- phase 1: stage ccw/cw msg rows + x rows as bf16, build mask bits ----
  for (int q = tid; q < NT * 17 * 64; q += 256) {
    int f4 = q & 63;
    int r = q >> 6;
    int node = r / 17, sub = r - node * 17;
    int ng = n0 + node; if (ng >= Ntot) ng = Ntot - 1;
    const float* src; u16* dst;
    if (sub < 8)       { src = ccw + (size_t)ng * 2048 + sub * 256;
                         dst = msgB + node * NODE_STRIDE + sub * MS; }
    else if (sub < 16) { src = cw + (size_t)ng * 2048 + (sub - 8) * 256;
                         dst = msgB + node * NODE_STRIDE + (sub + 2) * MS; }
    else               { src = bx + (size_t)ng * 256;
                         dst = xa + node * MS; }
    float4 v = *(const float4*)(src + f4 * 4);
    *(unsigned int*)(dst + f4 * 4)     = pk2bf(v.x, v.y);
    *(unsigned int*)(dst + f4 * 4 + 2) = pk2bf(v.z, v.w);
  }
  if (tid < NT) {
    int ng = n0 + tid; if (ng >= Ntot) ng = Ntot - 1;
    unsigned int mb = (1u << 8) | (1u << 9);        // self + parallel always on
    for (int k = 0; k < 8; ++k) {
      if (ccwm[(size_t)ng * 8 + k]) mb |= 1u << k;
      if (cwm[(size_t)ng * 8 + k])  mb |= 1u << (10 + k);
    }
    maskb[tid] = mb;
  }
  __syncthreads();

  // ---- phase 2: self_msg (slot 8) and kvec; waves split the 256 N cols ----
  {
    v8f a0 = zero8(), a1 = zero8();
    gemm_16x32(xa, MS, wSelfT, w * 32, lane, a0, a1);
    storeD_bf16(msgB + 8 * MS, NODE_STRIDE, w * 32, lane, a0, a1);
    v8f c0 = zero8(), c1 = zero8();
    gemm_16x32(msgB /*slot0 = ccw[:,0]*/, NODE_STRIDE, wKT, w * 32, lane, c0, c1);
    storeD_bf16(kkb, MS, w * 32, lane, c0, c1);
  }
  __syncthreads();

  // ---- phase 3: restage xa with gathered rows batch_x[pidx[n]] -------------
  for (int q = tid; q < NT * 64; q += 256) {
    int node = q >> 6, f4 = q & 63;
    int ng = n0 + node; if (ng >= Ntot) ng = Ntot - 1;
    int gi = pidx[ng];
    float4 v = *(const float4*)(bx + (size_t)gi * 256 + f4 * 4);
    u16* dst = xa + node * MS;
    *(unsigned int*)(dst + f4 * 4)     = pk2bf(v.x, v.y);
    *(unsigned int*)(dst + f4 * 4 + 2) = pk2bf(v.z, v.w);
  }
  __syncthreads();

  // ---- phase 4: parallel_msg (slot 9) + Z (one head per wave) --------------
  {
    v8f a0 = zero8(), a1 = zero8();
    gemm_16x32(xa, MS, wParT, w * 32, lane, a0, a1);
    storeD_bf16(msgB + 9 * MS, NODE_STRIDE, w * 32, lane, a0, a1);

    // Z[n,c,h] = sum_hd Wq[32h+hd,c] * kvec[n,32h+hd]  (K=32, single WMMA)
    const int h = w;
    v16bf af = ldA(kkb + 32 * h, MS, lane);
    int half = lane >> 4, cl = lane & 15;
#pragma unroll
    for (int ct = 0; ct < 16; ++ct) {
      v16bf b = ldBg(wQ + (size_t)(32 * h) * 256 + ct * 16, 256, lane);
      v8f az = zero8();
      az = wmma_bf16(af, b, az);
#pragma unroll
      for (int r = 0; r < 8; ++r) {
        int node = r + half * 8;
        int c = ct * 16 + cl;
        Zb[node * 2048 + c * 8 + h] = f2bf(az[r]);
      }
    }
  }
  __syncthreads();

  // ---- phase 5: scores[n,slot,h] via per-node WMMA (M=32 pad, N=16 pad) ----
  {
    const float scl = 0.1767766953f;   // 1/sqrt(32)
    for (int nn = w; nn < NT; nn += 8) {
      v8f s0 = zero8(), s1 = zero8();
      const u16* Ab = msgB + nn * NODE_STRIDE;
      const u16* Zn = Zb + nn * 2048;
#pragma unroll
      for (int kk = 0; kk < 256; kk += 32) {
        v16bf a0 = ldA(Ab + kk, MS, lane);
        v16bf a1 = ldA_g(Ab + 16 * MS + kk, MS, lane, 2);  // slots 16,17 only
        v16bf b  = ldB_half(Zn + (size_t)kk * 8, lane);
        s0 = wmma_bf16(a0, b, s0);
        s1 = wmma_bf16(a1, b, s1);
      }
      int half = lane >> 4, cl = lane & 15;
      if (cl < 8) {
#pragma unroll
        for (int r = 0; r < 8; ++r) {
          int slot0 = r + half * 8;
          scp[nn * 144 + slot0 * 8 + cl] = s0[r] * scl;
          int slot1 = 16 + r + half * 8;
          if (slot1 < KSLOT) scp[nn * 144 + slot1 * 8 + cl] = s1[r] * scl;
        }
      }
    }
  }
  __syncthreads();

  // ---- phase 6: masked softmax; emit p as bf16 in A-fragment layout --------
  if (tid < 128) {
    int n = tid >> 3, h = tid & 7;
    unsigned int mb = maskb[n];
    float s[KSLOT];
    float mx = -1e30f;
#pragma unroll
    for (int k = 0; k < KSLOT; ++k) {
      s[k] = scp[n * 144 + k * 8 + h];
      if ((mb >> k) & 1u) mx = fmaxf(mx, s[k]);
    }
    float sum = 0.f;
#pragma unroll
    for (int k = 0; k < KSLOT; ++k) {
      float e = ((mb >> k) & 1u) ? __expf(s[k] - mx) : 0.f;
      s[k] = e; sum += e;
    }
    float inv = 1.f / sum;                  // self slot always valid -> sum>0
    u16* pp = pb + n * PB_NS + h * PB_HS;
#pragma unroll
    for (int k = 0; k < KSLOT; ++k) pp[k] = f2bf(s[k] * inv);
#pragma unroll
    for (int k = KSLOT; k < 32; ++k) pp[k] = 0;     // zero K padding
  }
  __syncthreads();

  // ---- phase 7: ctx = p(16x32) x msg(32x256) per node, one WMMA per c-tile -
  // A rows = heads (8 valid), K = slots (18 valid); B rows = msg slots.
  {
    int half = lane >> 4, cl = lane & 15;
    for (int nn = w; nn < NT; nn += 8) {
      v16bf ap = ldA_g(pb + nn * PB_NS, PB_HS, lane, 8);
      const u16* Mb = msgB + nn * NODE_STRIDE;
#pragma unroll
      for (int ct = 0; ct < 16; ++ct) {
        v16bf b = ldB_rows(Mb + ct * 16, MS, lane, KSLOT);
        v8f d = zero8();
        d = wmma_bf16(ap, b, d);
        if (half == 0) {                       // D rows 0..7 = heads 0..7
#pragma unroll
          for (int r = 0; r < 8; ++r)
            ctxb[nn * CTX_STRIDE + r * MS + ct * 16 + cl] = f2bf(d[r]);
        }
      }
    }
  }
  __syncthreads();

  // ---- phase 8: per-head output projection: ctx_h(16x256) @ WvT cols -------
  {
    const int h = w;
    v8f a0 = zero8(), a1 = zero8();
    gemm_16x32(ctxb + h * MS, CTX_STRIDE, wVT, 32 * h, lane, a0, a1);
    storeD_bf16(xa /*reused as outv*/, MS, 32 * h, lane, a0, a1);
  }
  __syncthreads();

  // ---- phase 9: final projection, f32 stores to global ---------------------
  {
    v8f a0 = zero8(), a1 = zero8();
    gemm_16x32(xa, MS, wFinT, w * 32, lane, a0, a1);
    int half = lane >> 4, cl = lane & 15;
#pragma unroll
    for (int r = 0; r < 8; ++r) {
      int node = r + half * 8;
      int ng = n0 + node;
      if (ng < Ntot) {
        outp[(size_t)ng * 256 + w * 32 + cl]      = a0[r];
        outp[(size_t)ng * 256 + w * 32 + 16 + cl] = a1[r];
      }
    }
  }
}

// ============================================================================
extern "C" void kernel_launch(void* const* d_in, const int* in_sizes, int n_in,
                              void* d_out, int out_size, void* d_ws, size_t ws_size,
                              hipStream_t stream) {
  const float* bx   = (const float*)d_in[0];
  const int*   pidx = (const int*)d_in[1];
  const float* ccw  = (const float*)d_in[2];
  const unsigned char* ccwm = (const unsigned char*)d_in[3];
  const float* cw   = (const float*)d_in[4];
  const unsigned char* cwm  = (const unsigned char*)d_in[5];
  const float* wself = (const float*)d_in[6];
  const float* wpar  = (const float*)d_in[7];
  const float* wq    = (const float*)d_in[8];
  const float* wk    = (const float*)d_in[9];
  const float* wv    = (const float*)d_in[10];
  const float* wfin  = (const float*)d_in[11];
  float* outp = (float*)d_out;
  u16* wbf = (u16*)d_ws;                       // 6 * 128KB bf16 weight slabs

  int Ntot = in_sizes[0] / 256;

  wconv<<<dim3(256, 6), 256, 0, stream>>>(wself, wpar, wk, wq, wv, wfin, wbf);
  chienn_fused<<<dim3((Ntot + NT - 1) / NT), 256, 0, stream>>>(
      bx, pidx, ccw, ccwm, cw, cwm, wbf, outp, Ntot);
}
